// NSFModel_88630945120436
// MI455X (gfx1250) — compile-verified, matrix operand order checked by reading, not run on hardware
//
#include <hip/hip_runtime.h>
#include <hip/hip_bf16.h>

typedef __attribute__((ext_vector_type(16))) __bf16 v16bf;
typedef __attribute__((ext_vector_type(8)))  __bf16 v8bf;
typedef __attribute__((ext_vector_type(8)))  float  v8f;
typedef __attribute__((ext_vector_type(4)))  unsigned int v4u;
typedef __attribute__((ext_vector_type(8)))  int v8i;
typedef __attribute__((ext_vector_type(4)))  int v4i;

namespace {
constexpr int   kB      = 262144;
constexpr int   kE      = 64;
constexpr int   kH      = 128;
constexpr int   kK      = 32;
constexpr int   kL      = 6;
constexpr int   kNB     = 2;
constexpr int   kP      = 95;
constexpr float kTB     = 6.0f;
constexpr float kMinBin = 0.001f;
constexpr float kMinDer = 0.001f;

constexpr int KSTR   = 136;          // LDS row stride (bf16) for K=128 rows: 128 + 8 pad
constexpr int KSTR64 = 72;           // LDS row stride (bf16) for K=64  rows: 64 + 8 pad
constexpr int NWAVES = 8;
constexpr int TILE   = NWAVES * 16;  // 128 samples per workgroup
constexpr int PSTR   = 96;

constexpr int WBUF_BYTES = 128 * KSTR * 2;   // one weight buffer (34816 B); double buffered
constexpr int WSCR_BYTES = 16 * PSTR * 4;    // per-wave scratch: max(16*KSTR*2, 16*PSTR*4)
constexpr int SMEM_BYTES = 2 * WBUF_BYTES + NWAVES * WSCR_BYTES;   // ~116 KB

// bf16 weight stack layout in d_ws (element offsets), each matrix transposed [N][K] dense.
constexpr int WS_EW2    = 0;          // [64][64]
constexpr int WS_EW3    = 4096;       // [64][64]
constexpr int WS_LAYER0 = 8192;
constexpr int WS_LSTR   = 102400;     // per-layer stride
constexpr int WS_CTX    = 0;          // [128][64]
constexpr int WS_BLK    = 8192;       // + j*WS_BSTR
constexpr int WS_BSTR   = 40960;
constexpr int WS_WC     = 0;          // [128][64]
constexpr int WS_W1     = 8192;       // [128][128]
constexpr int WS_W2     = 24576;      // [128][128]
constexpr int WS_OUT    = 90112;      // [96][128] (col 95 zero-padded)
constexpr int WS_TOTAL  = WS_LAYER0 + kL * WS_LSTR;   // 622592 elems = ~1.2 MB
}  // namespace

// ---------------- device helpers ----------------

__device__ __forceinline__ float eluf(float x) {
  return x > 0.0f ? x : __expf(x) - 1.0f;
}
__device__ __forceinline__ float splusf(float x) {
  return x > 15.0f ? x : log1pf(__expf(x));
}

// Load one 16x32 bf16 fragment from an LDS matrix with row stride rstr (bf16 elems).
// CDNA5 16-bit 16x32 layout: lanes 0-15 kh=0, lanes 16-31 kh=8;
// per lane elements = K { kh..kh+7, 16+kh..16+kh+7 }  -> two ds_load_b128.
__device__ __forceinline__ v16bf ld_frag(const __bf16* mat, int row, int kc, int kh,
                                         int rstr) {
  const __bf16* p = mat + row * rstr + kc * 32 + kh;
  union { v16bf v; v8bf h[2]; } u;
  u.h[0] = *(const v8bf*)(p);
  u.h[1] = *(const v8bf*)(p + 16);
  return u.v;
}

__device__ __forceinline__ v8f wmma_b(v16bf a, v16bf b, v8f c) {
  return __builtin_amdgcn_wmma_f32_16x16x32_bf16(
      false, a, false, b, (short)0, c, false, false);
}

// acc[NT] += A(16 x KC*32, from abuf stride KSTR) x B(from wbuf stride bstr)
template <int KC, int NT>
__device__ __forceinline__ void gemm_lds(const __bf16* abuf, const __bf16* wbuf,
                                         int bstr, v8f* acc, int lo, int kh) {
#pragma unroll
  for (int kc = 0; kc < KC; ++kc) {
    v16bf a = ld_frag(abuf, lo, kc, kh, KSTR);
#pragma unroll
    for (int nt = 0; nt < NT; ++nt) {
      v16bf b = ld_frag(wbuf, nt * 16 + lo, kc, kh, bstr);
      acc[nt] = wmma_b(a, b, acc[nt]);
    }
  }
}

// Same but A (K=64) comes from persistent register fragments (the embedding).
template <int NT>
__device__ __forceinline__ void gemm_reg(const v16bf* aF, const __bf16* wbuf,
                                         int bstr, v8f* acc, int lo, int kh) {
#pragma unroll
  for (int kc = 0; kc < 2; ++kc) {
#pragma unroll
    for (int nt = 0; nt < NT; ++nt) {
      v16bf b = ld_frag(wbuf, nt * 16 + lo, kc, kh, bstr);
      acc[nt] = wmma_b(aF[kc], b, acc[nt]);
    }
  }
}

// Dump C-fragments (+bias, optional elu) into the per-wave LDS activation tile,
// bf16 row-major [16][KSTR].  C layout: VGPR r, lanes 0-15 -> (M=r), 16-31 -> (M=r+8).
template <int NT, bool ELU>
__device__ __forceinline__ void dump_act(const v8f* acc, const float* __restrict__ bias,
                                         __bf16* abuf, int lo, int hi) {
#pragma unroll
  for (int nt = 0; nt < NT; ++nt) {
    int col = nt * 16 + lo;
    float bv = bias ? bias[col] : 0.0f;
#pragma unroll
    for (int r = 0; r < 8; ++r) {
      float v = acc[nt][r] + bv;
      if (ELU) v = eluf(v);
      abuf[(r + 8 * hi) * KSTR + col] = (__bf16)v;
    }
  }
}

// Issue one TDM copy: dense bf16 [Nrows][Krow] at gsrc -> LDS at lds_off with
// +16B padding per row (pad_interval: 128B rows -> code 4, 256B rows -> code 5;
// pad_amount 4 dwords -> code 3).  2-D descriptor, remaining groups zero.
__device__ __forceinline__ void tdm_load(const __bf16* gsrc, unsigned int lds_off,
                                         int Nrows, int Krow) {
  unsigned long long ga = (unsigned long long)(uintptr_t)gsrc;
  v4u g0;
  g0.x = 1u;                                    // count=1 (valid user descriptor)
  g0.y = lds_off;                               // lds_addr (bytes)
  g0.z = (unsigned int)ga;                      // global_addr[31:0]
  g0.w = ((unsigned int)(ga >> 32) & 0x01FFFFFFu) | (2u << 30);  // addr[56:32] | type=2
  unsigned int padi = (Krow == 64) ? 4u : 5u;
  v8i g1;
  g1[0] = (int)((1u << 16)          // data_size = 2 bytes
              | (1u << 20)          // pad_enable
              | (padi << 22)        // pad_interval
              | (3u << 25));        // pad_amount = 4 dwords (16 B)
  g1[1] = (int)(((unsigned int)Krow & 0xFFFFu) << 16);              // tensor_dim0 lo16
  g1[2] = (int)(((unsigned int)Krow >> 16) |
                (((unsigned int)Nrows & 0xFFFFu) << 16));           // dim0 hi | dim1 lo
  g1[3] = (int)(((unsigned int)Nrows >> 16) |
                ((unsigned int)Krow << 16));                        // dim1 hi | tile_dim0
  g1[4] = (int)((unsigned int)Nrows & 0xFFFFu);                     // tile_dim1 | tile_dim2=0
  g1[5] = Krow;                                                     // tensor_dim0_stride lo32
  g1[6] = 0;
  g1[7] = 0;
  v4i g2 = {0, 0, 0, 0};
  v4i g3 = {0, 0, 0, 0};
  v8i g4 = {0, 0, 0, 0, 0, 0, 0, 0};
  __builtin_amdgcn_tensor_load_to_lds(g0, g1, g2, g3, g4, 0);
}

// ---------------- weight pre-conversion (fp32 -> bf16, transposed stack) ----------------

__global__ __launch_bounds__(256) void convert_weights(
    const float* __restrict__ eW2, const float* __restrict__ eW3,
    const float* __restrict__ ctx_W, const float* __restrict__ blk_Wc,
    const float* __restrict__ blk_W1, const float* __restrict__ blk_W2,
    const float* __restrict__ out_W, __bf16* __restrict__ ws) {
  int idx = blockIdx.x * 256 + threadIdx.x;
  if (idx >= WS_TOTAL) return;
  float v;
  if (idx < WS_EW3) {
    int n = idx >> 6, k = idx & 63;
    v = eW2[k * kE + n];
  } else if (idx < WS_LAYER0) {
    int t = idx - WS_EW3;
    int n = t >> 6, k = t & 63;
    v = eW3[k * kE + n];
  } else {
    int r = idx - WS_LAYER0;
    int l = r / WS_LSTR;
    int lr = r - l * WS_LSTR;
    if (lr < WS_BLK) {                               // ctx_W[l]: [64][128] -> [128][64]
      int n = lr >> 6, k = lr & 63;
      v = ctx_W[(l * kE + k) * kH + n];
    } else if (lr < WS_OUT) {
      int br = lr - WS_BLK;
      int j = br / WS_BSTR;
      int b = br - j * WS_BSTR;
      int bidx = l * kNB + j;
      if (b < WS_W1) {                               // Wc: [64][128] -> [128][64]
        int n = b >> 6, k = b & 63;
        v = blk_Wc[(bidx * kE + k) * kH + n];
      } else if (b < WS_W2) {                        // W1: [128][128] -> [128][128]
        int c = b - WS_W1;
        int n = c >> 7, k = c & 127;
        v = blk_W1[(bidx * kH + k) * kH + n];
      } else {                                       // W2
        int c = b - WS_W2;
        int n = c >> 7, k = c & 127;
        v = blk_W2[(bidx * kH + k) * kH + n];
      }
    } else {                                         // out_W: [128][95] -> [96][128]
      int c = lr - WS_OUT;
      int n = c >> 7, k = c & 127;
      v = (n < kP) ? out_W[(l * kH + k) * kP + n] : 0.0f;
    }
  }
  ws[idx] = (__bf16)v;
}

// ---------------- main kernel ----------------

__global__ __launch_bounds__(NWAVES * 32) void nsf_kernel(
    const float* __restrict__ y, const float* __restrict__ ctx,
    const float* __restrict__ eW1, const float* __restrict__ eb1,
    const float* __restrict__ eb2, const float* __restrict__ eb3,
    const float* __restrict__ init_b, const float* __restrict__ ctx_b,
    const float* __restrict__ blk_b1, const float* __restrict__ blk_b2,
    const float* __restrict__ blk_bc, const float* __restrict__ out_b,
    const __bf16* __restrict__ ws, float* __restrict__ out) {
  extern __shared__ __align__(16) char smem[];

  // Weight double-buffer pointer / TDM LDS offset, computed at runtime (no
  // constant-folded addrspacecast initializers).
#define WB(i)   ((__bf16*)(smem + (i) * WBUF_BYTES))
#define LOFF(i) (lds0 + (unsigned int)((i) * WBUF_BYTES))
  const unsigned int lds0 = (unsigned int)(uintptr_t)smem;  // low 32 bits = LDS offset

  const int tid  = threadIdx.x;
  const int lane = tid & 31;
  const int wave = tid >> 5;
  const int lo   = lane & 15;
  const int hi   = lane >> 4;
  const int kh   = hi * 8;

  char*   wscr = smem + 2 * WBUF_BYTES + wave * WSCR_BYTES;
  __bf16* abuf = (__bf16*)wscr;   // [16][KSTR] bf16 activation tile (A staging)
  float*  pbuf = (float*)wscr;    // [16][PSTR] f32 spline params (aliased in time)

  const int s0 = blockIdx.x * TILE + wave * 16;
  const bool issuer = (wave == 0);

  // Acquire the TDM result for the current buffer, publish to all waves.
#define ACQ()                                                 \
  do {                                                        \
    if (issuer) __builtin_amdgcn_s_wait_tensorcnt(0);         \
    __syncthreads();                                          \
  } while (0)

  int cur = 0;
  if (issuer) tdm_load(ws + WS_EW2, LOFF(0), 64, 64);     // prefetch eW2

  float z = 0.0f, lad = 0.0f;

  // ---- embedding layer 1 (K=CD=2): VALU, one sample per lane 0..15 (overlaps TDM) ----
  if (lane < 16) {
    int s = s0 + lane;
    z = y[s];
    float c0 = ctx[2 * s + 0];
    float c1 = ctx[2 * s + 1];
    for (int e = 0; e < kE; ++e) {
      float v = fmaf(c0, eW1[e], fmaf(c1, eW1[kE + e], eb1[e]));
      abuf[lane * KSTR + e] = (__bf16)eluf(v);
    }
  }

  // ---- embedding layers 2 & 3 ----
  ACQ();                                                   // eW2 in buffer 0
  if (issuer) tdm_load(ws + WS_EW3, LOFF(1), 64, 64);
  {
    v8f e2[4] = {};
    gemm_lds<2, 4>(abuf, WB(cur), KSTR64, e2, lo, kh);
    dump_act<4, true>(e2, eb2, abuf, lo, hi);
  }
  cur ^= 1;
  ACQ();                                                   // eW3 in buffer 1
  if (issuer) tdm_load(ws + WS_LAYER0 + WS_CTX, LOFF(cur ^ 1), 128, 64);
  {
    v8f e3[4] = {};
    gemm_lds<2, 4>(abuf, WB(cur), KSTR64, e3, lo, kh);
    dump_act<4, true>(e3, eb3, abuf, lo, hi);
  }
  cur ^= 1;
  v16bf embA[2];
  embA[0] = ld_frag(abuf, lo, 0, kh, KSTR);
  embA[1] = ld_frag(abuf, lo, 1, kh, KSTR);

  // ---- flow layers ----
  for (int l = 0; l < kL; ++l) {
    const int lbase = WS_LAYER0 + l * WS_LSTR;

    ACQ();                                                 // ctx_W[l]
    if (issuer) tdm_load(ws + lbase + WS_BLK + WS_WC, LOFF(cur ^ 1), 128, 64);
    v8f h[8] = {};
    gemm_reg<8>(embA, WB(cur), KSTR64, h, lo, kh);
    cur ^= 1;
#pragma unroll
    for (int nt = 0; nt < 8; ++nt) {
      int col = nt * 16 + lo;
      float bv = init_b[l * kH + col] + ctx_b[l * kH + col];
#pragma unroll
      for (int r = 0; r < 8; ++r) h[nt][r] += bv;
    }

    for (int j = 0; j < kNB; ++j) {
      const int bidx = l * kNB + j;
      const int bbase = lbase + WS_BLK + j * WS_BSTR;

      ACQ();                                               // Wc
      if (issuer) tdm_load(ws + bbase + WS_W1, LOFF(cur ^ 1), 128, 128);
      v8f g[8] = {};
      gemm_reg<8>(embA, WB(cur), KSTR64, g, lo, kh);
      cur ^= 1;
#pragma unroll
      for (int nt = 0; nt < 8; ++nt) {
        int col = nt * 16 + lo;
        float bc = blk_bc[bidx * kH + col];
#pragma unroll
        for (int r = 0; r < 8; ++r)
          g[nt][r] = 1.0f / (1.0f + __expf(-(g[nt][r] + bc)));
      }
      dump_act<8, true>(h, nullptr, abuf, lo, hi);         // elu(h) -> abuf

      ACQ();                                               // W1
      if (issuer) tdm_load(ws + bbase + WS_W2, LOFF(cur ^ 1), 128, 128);
      v8f t[8] = {};
      gemm_lds<4, 8>(abuf, WB(cur), KSTR, t, lo, kh);
      cur ^= 1;
      dump_act<8, true>(t, blk_b1 + bidx * kH, abuf, lo, hi);  // elu(t+b1) -> abuf

      ACQ();                                               // W2
      if (issuer) {
        if (j == 0)
          tdm_load(ws + lbase + WS_BLK + WS_BSTR + WS_WC, LOFF(cur ^ 1), 128, 64);
        else
          tdm_load(ws + lbase + WS_OUT, LOFF(cur ^ 1), 96, 128);
      }
#pragma unroll
      for (int nt = 0; nt < 8; ++nt) t[nt] = (v8f){};
      gemm_lds<4, 8>(abuf, WB(cur), KSTR, t, lo, kh);
      cur ^= 1;
#pragma unroll
      for (int nt = 0; nt < 8; ++nt) {                     // h += (t + b2) * gate
        int col = nt * 16 + lo;
        float b2 = blk_b2[bidx * kH + col];
#pragma unroll
        for (int r = 0; r < 8; ++r)
          h[nt][r] += (t[nt][r] + b2) * g[nt][r];
      }
    }

    // p = h @ out_W[l] + out_b[l]   (h, not elu(h))
    dump_act<8, false>(h, nullptr, abuf, lo, hi);
    ACQ();                                                 // out_W[l]
    if (issuer && l + 1 < kL)
      tdm_load(ws + WS_LAYER0 + (l + 1) * WS_LSTR + WS_CTX, LOFF(cur ^ 1), 128, 64);
    v8f p[6] = {};
    gemm_lds<4, 6>(abuf, WB(cur), KSTR, p, lo, kh);
    cur ^= 1;
#pragma unroll
    for (int nt = 0; nt < 6; ++nt) {
      int col = nt * 16 + lo;
      if (col < kP) {
        float ob = out_b[l * kP + col];
#pragma unroll
        for (int r = 0; r < 8; ++r)
          pbuf[(r + 8 * hi) * PSTR + col] = p[nt][r] + ob;
      }
    }

    // ---- rational-quadratic spline, one sample per lane 0..15 ----
    if (lane < 16) {
      const float* pr = pbuf + lane * PSTR;
      bool  inside = (z >= -kTB) && (z <= kTB);
      float yc = fminf(fmaxf(z, -kTB), kTB);
      const float invs = 0.08838834764831845f;  // 1/sqrt(H)

      float mw = -3.4e38f, mh = -3.4e38f;
      for (int i = 0; i < kK; ++i) {
        mw = fmaxf(mw, pr[i]);
        mh = fmaxf(mh, pr[kK + i]);
      }
      mw *= invs; mh *= invs;
      float sw = 0.0f, sh = 0.0f;
      for (int i = 0; i < kK; ++i) {
        sw += __expf(pr[i] * invs - mw);
        sh += __expf(pr[kK + i] * invs - mh);
      }
      const float cmul = 1.0f - kMinBin * (float)kK;
      const float aw = cmul / sw, ah = cmul / sh;

      float cw = -kTB, ch = -kTB;
      float cwk = -kTB, wk = 2.0f * kTB, chk = -kTB, hk = 2.0f * kTB;
      float dk = 1.0f, dk1 = 1.0f;
      for (int i = 0; i < kK; ++i) {
        float wi  = 2.0f * kTB * (kMinBin + aw * __expf(pr[i] * invs - mw));
        float hgt = 2.0f * kTB * (kMinBin + ah * __expf(pr[kK + i] * invs - mh));
        float cwn = (i == kK - 1) ? kTB : cw + wi;
        float chn = (i == kK - 1) ? kTB : ch + hgt;
        if (yc >= cw) {  // left edge of bin i; last true i == bin index
          cwk = cw; wk = cwn - cw; chk = ch; hk = chn - ch;
          dk  = (i == 0)      ? 1.0f : kMinDer + splusf(pr[2 * kK + i - 1]);
          dk1 = (i == kK - 1) ? 1.0f : kMinDer + splusf(pr[2 * kK + i]);
        }
        cw = cwn; ch = chn;
      }
      float sk  = hk / wk;
      float th  = (yc - cwk) / wk;
      float t1m = th * (1.0f - th);
      float num = hk * (sk * th * th + dk * t1m);
      float den = sk + (dk + dk1 - 2.0f * sk) * t1m;
      float omt = 1.0f - th;
      float dnum = sk * sk * (dk1 * th * th + 2.0f * sk * t1m + dk * omt * omt);
      float ladd = __logf(dnum) - 2.0f * __logf(den);
      if (inside) {
        z = chk + num / den;
        lad += ladd;
      }
    }
  }

  if (lane < 16) {
    int s = s0 + lane;
    out[s] = -0.5f * z * z - 0.91893853320467274178f + lad;  // 0.5*log(2*pi)
  }
#undef ACQ
#undef WB
#undef LOFF
}

// ---------------- launcher ----------------

extern "C" void kernel_launch(void* const* d_in, const int* in_sizes, int n_in,
                              void* d_out, int out_size, void* d_ws, size_t ws_size,
                              hipStream_t stream) {
  (void)in_sizes; (void)n_in; (void)out_size; (void)ws_size;
  const float* y      = (const float*)d_in[0];
  const float* ctx    = (const float*)d_in[1];
  const float* eW1    = (const float*)d_in[2];
  const float* eb1    = (const float*)d_in[3];
  const float* eW2    = (const float*)d_in[4];
  const float* eb2    = (const float*)d_in[5];
  const float* eW3    = (const float*)d_in[6];
  const float* eb3    = (const float*)d_in[7];
  const float* init_b = (const float*)d_in[8];
  const float* ctx_W  = (const float*)d_in[9];
  const float* ctx_b  = (const float*)d_in[10];
  const float* blk_W1 = (const float*)d_in[11];
  const float* blk_b1 = (const float*)d_in[12];
  const float* blk_W2 = (const float*)d_in[13];
  const float* blk_b2 = (const float*)d_in[14];
  const float* blk_Wc = (const float*)d_in[15];
  const float* blk_bc = (const float*)d_in[16];
  const float* out_W  = (const float*)d_in[17];
  const float* out_b  = (const float*)d_in[18];
  float* out = (float*)d_out;
  __bf16* ws = (__bf16*)d_ws;   // needs WS_TOTAL*2 ~ 1.22 MB of scratch

  convert_weights<<<dim3((WS_TOTAL + 255) / 256), dim3(256), 0, stream>>>(
      eW2, eW3, ctx_W, blk_Wc, blk_W1, blk_W2, out_W, ws);

  nsf_kernel<<<dim3(kB / TILE), dim3(NWAVES * 32), SMEM_BYTES, stream>>>(
      y, ctx, eW1, eb1, eb2, eb3, init_b, ctx_b, blk_b1, blk_b2, blk_bc, out_b,
      ws, out);
}
